// IEGNN_81750407512689
// MI455X (gfx1250) — matrix operand order repaired and checked by reading, not compiled, not run on hardware
//
#include <hip/hip_runtime.h>
#include <hip/hip_bf16.h>

typedef __bf16 bf16_t;
typedef __attribute__((ext_vector_type(16))) __bf16 v16bf;
typedef __attribute__((ext_vector_type(8)))  __bf16 v8bf;
typedef __attribute__((ext_vector_type(8)))  float  v8f;

#define HD      128
#define NNODES  50000
#define NEDGES  800000
#define NLAYERS 8
#define NGRAPH  100
#define NPER    500

#define KP_E    288      // 2*128+1+4 = 261 padded to 9 K-steps of 32
#define KP_I    256      // interaction MLP K (8 K-steps, no pad)
#define KP_N    416      // 3*128+16 = 400 padded to 13 K-steps of 32
#define EWAVES  8        // waves/block, edge kernel
#define NWAVES  8        // waves/block, node kernel

// fragment-swizzled weight slice sizes (bf16 elements)
#define EW1S    (9 * 4096)    // 36864
#define EW2S    (4 * 4096)    // 16384
#define IW1S    (8 * 4096)    // 32768
#define NW1S    (13 * 4096)   // 53248
#define WEDGE   (EW1S + EW2S + IW1S + EW2S)   // 102400 elems = 204800 B
#define WNODE   (NW1S + EW2S)                 // 69632  elems = 139264 B
// LDS offsets (bf16 elements) inside dynamic LDS
#define OFF_EW1 0
#define OFF_EW2 (EW1S)
#define OFF_IW1 (EW1S + EW2S)
#define OFF_IW2 (EW1S + EW2S + IW1S)
#define ESTAGE0 (WEDGE)                       // edge staging base
#define ESTRIDE (16 * KP_E + 16 * HD)         // 6656 elems / wave
#define OFF_NW1 0
#define OFF_NW2 (NW1S)
#define NSTAGE0 (WNODE)
#define NSTRIDE (16 * KP_N + 16 * HD)         // 8704 elems / wave

// ---------------------------------------------------------------------------
// CDNA5 async global->LDS copy (ASYNCcnt) via inline asm (portable across
// ROCm7.2 / amdgpu-toolchain, bypasses the arity-divergent TDM builtin).
// GVS mode: mem = SADDR(64) + VADDR(32-bit per-lane offset); VDST = per-lane
// LDS byte offset (flat LDS addresses truncate to the LDS offset).
__device__ __forceinline__ void async_copy_b128(unsigned lds_off,
                                                unsigned long long sbase,
                                                unsigned goff) {
  asm volatile("global_load_async_to_lds_b128 %0, %1, %2"
               :: "v"(lds_off), "v"(goff), "s"(sbase)
               : "memory");
}
__device__ __forceinline__ void wait_async_all() {
  asm volatile("s_wait_asynccnt 0x0" ::: "memory");
}
__device__ __forceinline__ unsigned lds_off_of(const void* p) {
  return (unsigned)(uintptr_t)p;   // LDS aperture: addr[31:0] == LDS offset
}

// ---------------------------------------------------------------------------
// A-fragment (16x32 bf16) from an LDS staging tile.
// Lanes 0-15 -> M=lane, K in {0..7,16..23}; lanes 16-31 -> M=lane-16,
// K in {8..15,24..31}. Two aligned 16B LDS loads per lane.
__device__ __forceinline__ v16bf lds_a_frag(const bf16_t* base, int stride, int ks) {
  const int lane = threadIdx.x & 31;
  const int m = lane & 15, half = lane >> 4;
  const bf16_t* rp = base + m * stride + ks * 32 + half * 8;
  v8bf lo = *(const v8bf*)(rp);
  v8bf hi = *(const v8bf*)(rp + 16);
  v16bf a;
#pragma unroll
  for (int i = 0; i < 8; ++i) { a[i] = lo[i]; a[8 + i] = hi[i]; }
  return a;
}

// B-fragment (32x16 bf16) from LDS-cached, fragment-swizzled weights:
// each lane's 16 values are contiguous (32B aligned).
__device__ __forceinline__ v16bf lds_b_frag(const bf16_t* w, int ks, int nt) {
  const int lane = threadIdx.x & 31;
  return ((const v16bf*)w)[(ks * 8 + nt) * 32 + lane];
}

template<int KSTEPS>
__device__ __forceinline__ void wave_gemm(const bf16_t* Ast, int stride,
                                          const bf16_t* Wlds, v8f acc[8]) {
  for (int ks = 0; ks < KSTEPS; ++ks) {
    v16bf a = lds_a_frag(Ast, stride, ks);
#pragma unroll
    for (int nt = 0; nt < 8; ++nt) {
      v16bf b = lds_b_frag(Wlds, ks, nt);
      acc[nt] = __builtin_amdgcn_wmma_f32_16x16x32_bf16(
          false, a, false, b, (short)0, acc[nt], false, false);
    }
  }
}

__device__ __forceinline__ void zero_acc(v8f acc[8]) {
#pragma unroll
  for (int nt = 0; nt < 8; ++nt)
#pragma unroll
    for (int v = 0; v < 8; ++v) acc[nt][v] = 0.0f;
}

// ---------------------------------------------------------------------------
// Weight fragment swizzle: f32 [K][128] -> bf16 fragment order
__global__ void __launch_bounds__(256)
swizzle_kernel(const float* __restrict__ src, int K, int ksteps,
               bf16_t* __restrict__ dst) {
  int t = blockIdx.x * blockDim.x + threadIdx.x;
  if (t >= ksteps * 4096) return;
  int e = t & 15, lane = (t >> 4) & 31, nt = (t >> 9) & 7, ks = t >> 12;
  int k = ks * 32 + (lane >> 4) * 8 + (e & 7) + (e >> 3) * 16;
  int n = nt * 16 + (lane & 15);
  float v = (k < K) ? src[(size_t)k * HD + n] : 0.0f;
  dst[t] = (bf16_t)v;
}

__global__ void __launch_bounds__(256)
embed_kernel(const float* __restrict__ hin, const float* __restrict__ w,
             const float* __restrict__ b, float* __restrict__ outF,
             bf16_t* __restrict__ outB) {
  int t = blockIdx.x * blockDim.x + threadIdx.x;
  if (t >= NNODES * HD) return;
  int node = t >> 7, f = t & 127;
  float acc = b[f];
#pragma unroll
  for (int k = 0; k < 16; ++k) acc += hin[node * 16 + k] * w[k * HD + f];
  if (outF) outF[t] = acc;
  outB[t] = (bf16_t)acc;
}

__global__ void __launch_bounds__(256)
radial_kernel(const int* __restrict__ edges, const float* __restrict__ coord,
              float* __restrict__ radial) {
  int e = blockIdx.x * blockDim.x + threadIdx.x;
  if (e >= NEDGES) return;
  int r = edges[e], c = edges[NEDGES + e];
  float dx = coord[r * 3 + 0] - coord[c * 3 + 0];
  float dy = coord[r * 3 + 1] - coord[c * 3 + 1];
  float dz = coord[r * 3 + 2] - coord[c * 3 + 2];
  radial[e] = dx * dx + dy * dy + dz * dz;
}

__global__ void __launch_bounds__(256)
zero2_kernel(float* __restrict__ a, float* __restrict__ b, int n) {
  int t = blockIdx.x * blockDim.x + threadIdx.x;
  if (t < n) { a[t] = 0.0f; b[t] = 0.0f; }
}

// ---------------------------------------------------------------------------
// Edge kernel: block stages all 4 edge/interaction weight matrices into LDS
// via async DMA; each wave then runs a 16-edge tile through both MLPs with
// async-gathered A tiles, WMMA, relu, and atomic scatter-add.
__global__ void __launch_bounds__(EWAVES * 32)
edge_kernel(const int* __restrict__ rows, const int* __restrict__ cols,
            const int* __restrict__ irows, const int* __restrict__ icols,
            const float* __restrict__ radial, const float* __restrict__ eattr,
            const bf16_t* __restrict__ xbf, const bf16_t* __restrict__ xibf,
            const bf16_t* __restrict__ wedge,
            const float* __restrict__ eb1, const float* __restrict__ eb2,
            const float* __restrict__ ib1, const float* __restrict__ ib2,
            float* __restrict__ agg, float* __restrict__ iagg) {
  extern __shared__ bf16_t smem[];

  // ---- cooperative async weight stage: WEDGE*2 bytes, 16B chunks ----
  {
    unsigned lds0 = lds_off_of(smem);
    unsigned long long src = (unsigned long long)wedge;
    for (int i = 0; i < (WEDGE * 2 / 16) / (EWAVES * 32); ++i) {
      unsigned c = (threadIdx.x + i * (EWAVES * 32)) * 16u;
      async_copy_b128(lds0 + c, src, c);
    }
    wait_async_all();
    __syncthreads();
  }

  const int wave = threadIdx.x >> 5;
  const int lane = threadIdx.x & 31;
  bf16_t* Ast = smem + ESTAGE0 + wave * ESTRIDE;
  bf16_t* Hst = Ast + 16 * KP_E;
  const int tile = blockIdx.x * EWAVES + wave;
  if (tile * 16 >= NEDGES) return;            // wave-uniform guard
  const int er = lane & 15, hf = lane >> 4;
  const int cl = er;
  const int eidx = tile * 16 + er;
  const unsigned astOff = lds_off_of(Ast);
  const unsigned long long xb = (unsigned long long)xbf;

  v8f acc[8];

  // ---- async gather edge-MLP input: [x[row] | x[col] | radial | eattr | pad]
  {
    int r = rows[eidx], c = cols[eidx];
    unsigned gr = (unsigned)(((size_t)r * HD + hf * 64) * sizeof(bf16_t));
    unsigned gc = (unsigned)(((size_t)c * HD + hf * 64) * sizeof(bf16_t));
    unsigned dr = astOff + (unsigned)((er * KP_E + hf * 64) * 2);
    unsigned dc = astOff + (unsigned)((er * KP_E + HD + hf * 64) * 2);
#pragma unroll
    for (int i = 0; i < 8; ++i) {
      async_copy_b128(dr + i * 16, xb, gr + i * 16);
      async_copy_b128(dc + i * 16, xb, gc + i * 16);
    }
    if (hf == 0) {
      Ast[er * KP_E + 256] = (bf16_t)radial[eidx];
#pragma unroll
      for (int j = 0; j < 4; ++j)
        Ast[er * KP_E + 257 + j] = (bf16_t)eattr[eidx * 4 + j];
      for (int j = 261; j < KP_E; ++j) Ast[er * KP_E + j] = (bf16_t)0.0f;
    }
    wait_async_all();
  }

  // ---- edge MLP ----
  zero_acc(acc);
  wave_gemm<KP_E / 32>(Ast, KP_E, smem + OFF_EW1, acc);
#pragma unroll
  for (int nt = 0; nt < 8; ++nt) {
    float b = eb1[nt * 16 + cl];
#pragma unroll
    for (int v = 0; v < 8; ++v) {
      float x = acc[nt][v] + b;
      Hst[(v + hf * 8) * HD + nt * 16 + cl] = (bf16_t)(x > 0.0f ? x : 0.0f);
    }
  }
  zero_acc(acc);
  wave_gemm<HD / 32>(Hst, HD, smem + OFF_EW2, acc);
  {
    int nrow[8];
#pragma unroll
    for (int v = 0; v < 8; ++v) nrow[v] = rows[tile * 16 + v + hf * 8];
#pragma unroll
    for (int nt = 0; nt < 8; ++nt) {
      float b = eb2[nt * 16 + cl];
#pragma unroll
      for (int v = 0; v < 8; ++v) {
        float x = acc[nt][v] + b;
        x = x > 0.0f ? x : 0.0f;
        atomicAdd(&agg[(size_t)nrow[v] * HD + nt * 16 + cl], x);
      }
    }
  }

  // ---- interaction MLP: [x[irow] | xi[icol]] (K=256) ----
  {
    int r = irows[eidx], c = icols[eidx];
    const unsigned long long xib = (unsigned long long)xibf;
    unsigned gr = (unsigned)(((size_t)r * HD + hf * 64) * sizeof(bf16_t));
    unsigned gc = (unsigned)(((size_t)c * HD + hf * 64) * sizeof(bf16_t));
    unsigned dr = astOff + (unsigned)((er * KP_E + hf * 64) * 2);
    unsigned dc = astOff + (unsigned)((er * KP_E + HD + hf * 64) * 2);
#pragma unroll
    for (int i = 0; i < 8; ++i) {
      async_copy_b128(dr + i * 16, xb, gr + i * 16);
      async_copy_b128(dc + i * 16, xib, gc + i * 16);
    }
    wait_async_all();
  }
  zero_acc(acc);
  wave_gemm<KP_I / 32>(Ast, KP_E, smem + OFF_IW1, acc);
#pragma unroll
  for (int nt = 0; nt < 8; ++nt) {
    float b = ib1[nt * 16 + cl];
#pragma unroll
    for (int v = 0; v < 8; ++v) {
      float x = acc[nt][v] + b;
      Hst[(v + hf * 8) * HD + nt * 16 + cl] = (bf16_t)(x > 0.0f ? x : 0.0f);
    }
  }
  zero_acc(acc);
  wave_gemm<HD / 32>(Hst, HD, smem + OFF_IW2, acc);
  {
    int nrow[8];
#pragma unroll
    for (int v = 0; v < 8; ++v) nrow[v] = irows[tile * 16 + v + hf * 8];
#pragma unroll
    for (int nt = 0; nt < 8; ++nt) {
      float b = ib2[nt * 16 + cl];
#pragma unroll
      for (int v = 0; v < 8; ++v) {
        float x = acc[nt][v] + b;
        x = x > 0.0f ? x : 0.0f;
        atomicAdd(&iagg[(size_t)nrow[v] * HD + nt * 16 + cl], x);
      }
    }
  }
}

// ---------------------------------------------------------------------------
// Node kernel: nin = [x | agg | iagg | node_attr] -> 2-stage WMMA MLP.
__global__ void __launch_bounds__(NWAVES * 32)
node_kernel(const bf16_t* __restrict__ xbf, const float* __restrict__ agg,
            const float* __restrict__ iagg, const float* __restrict__ nattr,
            const bf16_t* __restrict__ wnode,
            const float* __restrict__ nb1, const float* __restrict__ nb2,
            float* __restrict__ outF, bf16_t* __restrict__ outB) {
  extern __shared__ bf16_t smem[];

  // ---- cooperative async weight stage: WNODE*2 bytes ----
  {
    unsigned lds0 = lds_off_of(smem);
    unsigned long long src = (unsigned long long)wnode;
    for (int i = 0; i < (WNODE * 2 / 16) / (NWAVES * 32); ++i) {
      unsigned c = (threadIdx.x + i * (NWAVES * 32)) * 16u;
      async_copy_b128(lds0 + c, src, c);
    }
    wait_async_all();
    __syncthreads();
  }

  const int wave = threadIdx.x >> 5;
  const int lane = threadIdx.x & 31;
  bf16_t* Ast = smem + NSTAGE0 + wave * NSTRIDE;
  bf16_t* Hst = Ast + 16 * KP_N;
  const int tile = blockIdx.x * NWAVES + wave;
  if (tile * 16 >= NNODES) return;            // wave-uniform guard
  const int er = lane & 15, hf = lane >> 4;
  const int cl = er;
  const int node = tile * 16 + er;

  // stage: x(async) | agg(cvt) | iagg(cvt) | nattr | pad
  {
    unsigned astOff = lds_off_of(Ast);
    unsigned gx = (unsigned)(((size_t)node * HD + hf * 64) * sizeof(bf16_t));
    unsigned dx = astOff + (unsigned)((er * KP_N + hf * 64) * 2);
#pragma unroll
    for (int i = 0; i < 8; ++i)
      async_copy_b128(dx + i * 16, (unsigned long long)xbf, gx + i * 16);
    for (int j = 0; j < 64; ++j) {
      Ast[er * KP_N + 128 + hf * 64 + j] = (bf16_t)agg[(size_t)node * HD + hf * 64 + j];
      Ast[er * KP_N + 256 + hf * 64 + j] = (bf16_t)iagg[(size_t)node * HD + hf * 64 + j];
    }
    if (hf == 0) {
#pragma unroll
      for (int j = 0; j < 16; ++j)
        Ast[er * KP_N + 384 + j] = (bf16_t)nattr[node * 16 + j];
#pragma unroll
      for (int j = 400; j < KP_N; ++j) Ast[er * KP_N + j] = (bf16_t)0.0f;
    }
    wait_async_all();
  }

  v8f acc[8];
  zero_acc(acc);
  wave_gemm<KP_N / 32>(Ast, KP_N, smem + OFF_NW1, acc);
#pragma unroll
  for (int nt = 0; nt < 8; ++nt) {
    float b = nb1[nt * 16 + cl];
#pragma unroll
    for (int v = 0; v < 8; ++v) {
      float x = acc[nt][v] + b;
      Hst[(v + hf * 8) * HD + nt * 16 + cl] = (bf16_t)(x > 0.0f ? x : 0.0f);
    }
  }
  zero_acc(acc);
  wave_gemm<HD / 32>(Hst, HD, smem + OFF_NW2, acc);
#pragma unroll
  for (int nt = 0; nt < 8; ++nt) {
    float b = nb2[nt * 16 + cl];
#pragma unroll
    for (int v = 0; v < 8; ++v) {
      float x = acc[nt][v] + b;       // no relu on node output
      size_t n2 = (size_t)(tile * 16 + v + hf * 8) * HD + nt * 16 + cl;
      outF[n2] = x;
      outB[n2] = (bf16_t)x;
    }
  }
}

// ---------------------------------------------------------------------------
__global__ void __launch_bounds__(HD)
pool_kernel(const float* __restrict__ x, const float* __restrict__ mask,
            float* __restrict__ pooled) {
  int b = blockIdx.x, f = threadIdx.x;
  float s = 0.0f;
  for (int i = 0; i < NPER; ++i) {
    int n = b * NPER + i;
    s += x[(size_t)n * HD + f] * mask[n];
  }
  pooled[b * HD + f] = s;
}

__global__ void __launch_bounds__(256)
dec_kernel(const float* __restrict__ pooled, const float* __restrict__ dw1,
           const float* __restrict__ db1, const float* __restrict__ dw2,
           const float* __restrict__ db2, float* __restrict__ out) {
  __shared__ float red[256];
  int b = blockIdx.x, j = threadIdx.x;
  float h = db1[j];
  for (int k = 0; k < HD; ++k) h += pooled[b * HD + k] * dw1[k * 256 + j];
  h = h > 0.0f ? h : 0.0f;
  red[j] = h * dw2[j];
  __syncthreads();
  for (int s = 128; s > 0; s >>= 1) {
    if (j < s) red[j] += red[j + s];
    __syncthreads();
  }
  if (j == 0) out[b] = red[0] + db2[0];
}

// ---------------------------------------------------------------------------
extern "C" void kernel_launch(void* const* d_in, const int* in_sizes, int n_in,
                              void* d_out, int out_size, void* d_ws, size_t ws_size,
                              hipStream_t stream) {
  const float* h_in  = (const float*)d_in[0];
  const int*   edges = (const int*)d_in[1];
  const float* eattr = (const float*)d_in[2];
  const float* nattr = (const float*)d_in[3];
  const float* coord = (const float*)d_in[4];
  const float* nmask = (const float*)d_in[6];
  const float* int_h = (const float*)d_in[7];
  const int*   iedges= (const int*)d_in[8];
  const float* emb_w = (const float*)d_in[9];
  const float* emb_b = (const float*)d_in[10];
  const float* ew1 = (const float*)d_in[11]; const float* eb1 = (const float*)d_in[12];
  const float* ew2 = (const float*)d_in[13]; const float* eb2 = (const float*)d_in[14];
  const float* iw1 = (const float*)d_in[15]; const float* ib1 = (const float*)d_in[16];
  const float* iw2 = (const float*)d_in[17]; const float* ib2 = (const float*)d_in[18];
  const float* nw1 = (const float*)d_in[19]; const float* nb1 = (const float*)d_in[20];
  const float* nw2 = (const float*)d_in[21]; const float* nb2 = (const float*)d_in[22];
  const float* dw1 = (const float*)d_in[23]; const float* db1 = (const float*)d_in[24];
  const float* dw2 = (const float*)d_in[25]; const float* db2 = (const float*)d_in[26];
  float* out = (float*)d_out;

  // ---- workspace carve-out (256B aligned) ----
  char* ws = (char*)d_ws;
  size_t off = 0;
  auto carve = [&](size_t bytes) -> char* {
    char* p = ws + off;
    off += (bytes + 255) & ~(size_t)255;
    return p;
  };
  float*  xA     = (float*)carve(sizeof(float) * (size_t)NNODES * HD);
  float*  xB     = (float*)carve(sizeof(float) * (size_t)NNODES * HD);
  float*  agg    = (float*)carve(sizeof(float) * (size_t)NNODES * HD);
  float*  iagg   = (float*)carve(sizeof(float) * (size_t)NNODES * HD);
  float*  radial = (float*)carve(sizeof(float) * (size_t)NEDGES);
  float*  pooled = (float*)carve(sizeof(float) * NGRAPH * HD);
  bf16_t* xAb    = (bf16_t*)carve(sizeof(bf16_t) * (size_t)NNODES * HD);
  bf16_t* xBb    = (bf16_t*)carve(sizeof(bf16_t) * (size_t)NNODES * HD);
  bf16_t* xib    = (bf16_t*)carve(sizeof(bf16_t) * (size_t)NNODES * HD);
  bf16_t* wedge  = (bf16_t*)carve(sizeof(bf16_t) * (size_t)NLAYERS * WEDGE);
  bf16_t* wnode  = (bf16_t*)carve(sizeof(bf16_t) * (size_t)NLAYERS * WNODE);
  (void)ws_size; (void)in_sizes; (void)n_in; (void)out_size;

  // ---- weight swizzle (f32 -> bf16 fragment order), packed per layer ----
  for (int l = 0; l < NLAYERS; ++l) {
    bf16_t* we = wedge + (size_t)l * WEDGE;
    bf16_t* wn = wnode + (size_t)l * WNODE;
    swizzle_kernel<<<(EW1S + 255) / 256, 256, 0, stream>>>(ew1 + (size_t)l * 261 * HD, 261, KP_E / 32, we + OFF_EW1);
    swizzle_kernel<<<(EW2S + 255) / 256, 256, 0, stream>>>(ew2 + (size_t)l * HD * HD, HD, HD / 32, we + OFF_EW2);
    swizzle_kernel<<<(IW1S + 255) / 256, 256, 0, stream>>>(iw1 + (size_t)l * 256 * HD, 256, KP_I / 32, we + OFF_IW1);
    swizzle_kernel<<<(EW2S + 255) / 256, 256, 0, stream>>>(iw2 + (size_t)l * HD * HD, HD, HD / 32, we + OFF_IW2);
    swizzle_kernel<<<(NW1S + 255) / 256, 256, 0, stream>>>(nw1 + (size_t)l * 400 * HD, 400, KP_N / 32, wn + OFF_NW1);
    swizzle_kernel<<<(EW2S + 255) / 256, 256, 0, stream>>>(nw2 + (size_t)l * HD * HD, HD, HD / 32, wn + OFF_NW2);
  }

  // ---- embeddings + radial ----
  const int embTot = NNODES * HD;
  embed_kernel<<<(embTot + 255) / 256, 256, 0, stream>>>(h_in, emb_w, emb_b, xA, xAb);
  embed_kernel<<<(embTot + 255) / 256, 256, 0, stream>>>(int_h, emb_w, emb_b, nullptr, xib);
  radial_kernel<<<(NEDGES + 255) / 256, 256, 0, stream>>>(edges, coord, radial);

  // ---- layers ----
  const size_t edgeLds = (size_t)(WEDGE + EWAVES * ESTRIDE) * sizeof(bf16_t);  // 311296 B
  const size_t nodeLds = (size_t)(WNODE + NWAVES * NSTRIDE) * sizeof(bf16_t);  // 278528 B
  hipFuncSetAttribute((const void*)edge_kernel,
                      hipFuncAttributeMaxDynamicSharedMemorySize, (int)edgeLds);
  hipFuncSetAttribute((const void*)node_kernel,
                      hipFuncAttributeMaxDynamicSharedMemorySize, (int)nodeLds);
  const int edgeBlocks = (NEDGES / 16) / EWAVES;               // 6250
  const int nodeBlocks = (NNODES / 16 + NWAVES - 1) / NWAVES;  // 391
  float* curF = xA; bf16_t* curB = xAb;
  float* nxtF = xB; bf16_t* nxtB = xBb;
  for (int l = 0; l < NLAYERS; ++l) {
    zero2_kernel<<<(NNODES * HD + 255) / 256, 256, 0, stream>>>(agg, iagg, NNODES * HD);
    edge_kernel<<<edgeBlocks, EWAVES * 32, edgeLds, stream>>>(
        edges, edges + NEDGES, iedges, iedges + NEDGES, radial, eattr,
        curB, xib, wedge + (size_t)l * WEDGE,
        eb1 + l * HD, eb2 + l * HD, ib1 + l * HD, ib2 + l * HD,
        agg, iagg);
    node_kernel<<<nodeBlocks, NWAVES * 32, nodeLds, stream>>>(
        curB, agg, iagg, nattr, wnode + (size_t)l * WNODE,
        nb1 + l * HD, nb2 + l * HD, nxtF, nxtB);
    float* tf = curF; curF = nxtF; nxtF = tf;
    bf16_t* tb = curB; curB = nxtB; nxtB = tb;
  }

  // ---- pooling + decoder ----
  pool_kernel<<<NGRAPH, HD, 0, stream>>>(curF, nmask, pooled);
  dec_kernel<<<NGRAPH, 256, 0, stream>>>(pooled, dw1, db1, dw2, db2, out);
}
